// WindowedMaskedAttention_82712480186752
// MI455X (gfx1250) — matrix operand order repaired and checked by reading, not compiled, not run on hardware
//
#include <hip/hip_runtime.h>
#include <hip/hip_bf16.h>

// ---------------------------------------------------------------------------
// Shapes (fixed by the reference)
// ---------------------------------------------------------------------------
#define Bq   2
#define Pq   32
#define Nq   512
#define Cq   384
#define Hq   12
#define Dq   32
#define TOK  (Bq * Pq * Nq)          // 32768 tokens
#define C3   (3 * Cq)                // 1152
#define NHEADS (Bq * Pq * Hq)        // 768 (b,p,h) heads

typedef __attribute__((ext_vector_type(16))) _Float16 v16h;
typedef __attribute__((ext_vector_type(8)))  _Float16 v8h;
typedef __attribute__((ext_vector_type(4)))  _Float16 v4h;
typedef __attribute__((ext_vector_type(8)))  float    v8f;
typedef __attribute__((ext_vector_type(4)))  float    v4f;

// ---------------------------------------------------------------------------
// WMMA helpers  (CDNA5: V_WMMA_F32_16X16X32_F16, wave32)
// ---------------------------------------------------------------------------
__device__ __forceinline__ v8f wmma_f16(v16h a, v16h b, v8f c) {
  // 8 args: (neg_a, A, neg_b, B, c_mod, C, reuse_a, reuse_b)
  return __builtin_amdgcn_wmma_f32_16x16x32_f16(false, a, false, b,
                                                (short)0, c, false, false);
}

__device__ __forceinline__ v16h cat8(v8h lo, v8h hi) {
  return __builtin_shufflevector(lo, hi, 0, 1, 2, 3, 4, 5, 6, 7,
                                 8, 9, 10, 11, 12, 13, 14, 15);
}

// A-fragment of a row-major [16 x 32] f16 tile.
// ISA layout: lane L holds row m = L%16; VGPR0-3 hold K = kb..kb+7,
// VGPR4-7 hold K = 16+kb..16+kb+7, kb = (L/16)*8.
__device__ __forceinline__ v16h load_a_frag(const _Float16* __restrict__ p,
                                            int ld, int row0, int k0) {
  const int l  = threadIdx.x & 31;
  const int m  = l & 15;
  const int kb = (l >> 4) * 8;
  const _Float16* r = p + (size_t)(row0 + m) * ld + (k0 + kb);
  v8h lo = *(const v8h*)(r);        // K = k0+kb .. +7
  v8h hi = *(const v8h*)(r + 16);   // K = k0+16+kb .. +7
  return cat8(lo, hi);
}

// B-fragment for B = [32(K) x 16(N)], supplied as row-major B^T [16 x 32]
// (i.e. p[n][k]).  Layout matches A: lane L holds column n = L%16;
// VGPR0-3 hold K = kb..kb+7, VGPR4-7 hold K = 16+kb..16+kb+7, kb = (L/16)*8.
__device__ __forceinline__ v16h load_b_frag(const _Float16* __restrict__ p,
                                            int ld, int row0, int k0) {
  const int l  = threadIdx.x & 31;
  const int n  = l & 15;
  const int kb = (l >> 4) * 8;
  const _Float16* r = p + (size_t)(row0 + n) * ld + (k0 + kb);
  v8h lo = *(const v8h*)(r);        // K = k0+kb .. +7
  v8h hi = *(const v8h*)(r + 16);   // K = k0+16+kb .. +7
  return cat8(lo, hi);
}

// ---------------------------------------------------------------------------
// Stage 0: f32 -> f16 conversion (x, w_qkv, w_proj), 4-wide
// ---------------------------------------------------------------------------
__global__ void cvt_f32_f16(const float* __restrict__ in,
                            _Float16* __restrict__ out, int n4) {
  int i = blockIdx.x * blockDim.x + threadIdx.x;
  const int stride = gridDim.x * blockDim.x;
  for (; i < n4; i += stride) {
    v4f x = ((const v4f*)in)[i];
    v4h y;
    #pragma unroll
    for (int j = 0; j < 4; ++j) y[j] = (_Float16)x[j];
    ((v4h*)out)[i] = y;
  }
}

// ---------------------------------------------------------------------------
// Stage 1: QKV projection.  out[t,o] = sum_c x[t,c]*w_qkv[o,c] + b_qkv[o]
// One wave per 32(M) x 64(N) tile; 8 WMMAs per 32-K step, K=384.
// Epilogue scatters into Q[bph][n][d] (pre-scaled), K[bph][n][d], VT[bph][d][n].
// ---------------------------------------------------------------------------
__global__ void __launch_bounds__(32)
gemm_qkv_kernel(const _Float16* __restrict__ xh,
                const _Float16* __restrict__ wh,
                const float*   __restrict__ bias,
                _Float16* __restrict__ qd,
                _Float16* __restrict__ kd,
                _Float16* __restrict__ vtd) {
  const int row0 = blockIdx.x * 32;   // token tile
  const int col0 = blockIdx.y * 64;   // output-feature tile
  v8f acc[2][4];
  #pragma unroll
  for (int i = 0; i < 2; ++i)
    #pragma unroll
    for (int j = 0; j < 4; ++j) acc[i][j] = (v8f){};

  for (int kk = 0; kk < Cq; kk += 32) {
    v16h a0 = load_a_frag(xh, Cq, row0,      kk);
    v16h a1 = load_a_frag(xh, Cq, row0 + 16, kk);
    v16h b0 = load_b_frag(wh, Cq, col0,      kk);
    v16h b1 = load_b_frag(wh, Cq, col0 + 16, kk);
    v16h b2 = load_b_frag(wh, Cq, col0 + 32, kk);
    v16h b3 = load_b_frag(wh, Cq, col0 + 48, kk);
    acc[0][0] = wmma_f16(a0, b0, acc[0][0]);
    acc[0][1] = wmma_f16(a0, b1, acc[0][1]);
    acc[0][2] = wmma_f16(a0, b2, acc[0][2]);
    acc[0][3] = wmma_f16(a0, b3, acc[0][3]);
    acc[1][0] = wmma_f16(a1, b0, acc[1][0]);
    acc[1][1] = wmma_f16(a1, b1, acc[1][1]);
    acc[1][2] = wmma_f16(a1, b2, acc[1][2]);
    acc[1][3] = wmma_f16(a1, b3, acc[1][3]);
  }

  const float qscale = 0.17677669529663687f;  // D^-0.5
  const int lane = threadIdx.x & 31;
  const int nl = lane & 15, hb = lane >> 4;   // C/D layout: n=nl, m=r+8*hb
  #pragma unroll
  for (int i = 0; i < 2; ++i) {
    #pragma unroll
    for (int j = 0; j < 4; ++j) {
      const int o  = col0 + j * 16 + nl;      // o = s*384 + h*32 + d
      const int s  = o / Cq;
      const int hd = o % Cq;
      const int h  = hd >> 5, d = hd & 31;
      const float bv = bias[o];
      #pragma unroll
      for (int r = 0; r < 8; ++r) {
        const int t  = row0 + i * 16 + hb * 8 + r;   // token = bp*512 + n
        const int bp = t >> 9, n = t & 511;
        const size_t bph = (size_t)bp * Hq + h;
        const float val = acc[i][j][r] + bv;
        if (s == 0)      qd [(bph * Nq + n) * Dq + d] = (_Float16)(val * qscale);
        else if (s == 1) kd [(bph * Nq + n) * Dq + d] = (_Float16)val;
        else             vtd[(bph * Dq + d) * Nq + n] = (_Float16)val; // V^T
      }
    }
  }
}

// ---------------------------------------------------------------------------
// Stage 2: flash attention.  One block = (head, 128-row slab); 8 waves,
// each wave owns 16 query rows, streams 32 keys/step with online softmax.
// ---------------------------------------------------------------------------
__global__ void __launch_bounds__(256)
attn_kernel(const _Float16* __restrict__ qd,
            const _Float16* __restrict__ kd,
            const _Float16* __restrict__ vtd,
            const float*   __restrict__ pos,
            _Float16* __restrict__ ao) {
  __shared__ __align__(16) _Float16 pbuf[8 * 16 * 32];  // 8 KB, 1 KB/wave

  const int wave = threadIdx.x >> 5;
  const int lane = threadIdx.x & 31;
  const int nl = lane & 15, hb = lane >> 4;
  const int bph    = blockIdx.x >> 2;   // 0..767
  const int rowblk = blockIdx.x & 3;
  const int h  = bph % Hq;
  const int bp = bph / Hq;
  const int row0 = rowblk * 128 + wave * 16;

  const _Float16* Q  = qd  + (size_t)bph * Nq * Dq;
  const _Float16* K  = kd  + (size_t)bph * Nq * Dq;
  const _Float16* VT = vtd + (size_t)bph * Nq * Dq;
  const float*    PS = pos + (size_t)h * Nq * Nq;
  _Float16* pb = pbuf + wave * (16 * 32);

  const v16h qf = load_a_frag(Q, Dq, row0, 0);   // [16 rows x 32 d], pre-scaled

  float mrow[8], lrow[8];
  #pragma unroll
  for (int r = 0; r < 8; ++r) { mrow[r] = -3.0e38f; lrow[r] = 0.0f; }
  v8f o0 = (v8f){};   // output cols 0..15
  v8f o1 = (v8f){};   // output cols 16..31

  for (int j0 = 0; j0 < Nq; j0 += 32) {
    // S = Q * K^T  (two 16-key tiles)
    v16h kf0 = load_b_frag(K, Dq, j0,      0);
    v16h kf1 = load_b_frag(K, Dq, j0 + 16, 0);
    v8f z = (v8f){};
    v8f s0 = wmma_f16(qf, kf0, z);
    v8f s1 = wmma_f16(qf, kf1, z);

    float mnew[8];
    #pragma unroll
    for (int r = 0; r < 8; ++r) {
      const int qi = row0 + hb * 8 + r;
      s0[r] += PS[(size_t)qi * Nq + j0 + nl];
      s1[r] += PS[(size_t)qi * Nq + j0 + 16 + nl];
      mnew[r] = fmaxf(s0[r], s1[r]);
    }
    // row max across the 16 lanes of this half (rows r+8*hb)
    #pragma unroll
    for (int off = 1; off < 16; off <<= 1)
      #pragma unroll
      for (int r = 0; r < 8; ++r)
        mnew[r] = fmaxf(mnew[r], __shfl_xor(mnew[r], off, 32));

    #pragma unroll
    for (int r = 0; r < 8; ++r) {
      const float mprev = mrow[r];
      const float mcur  = fmaxf(mprev, mnew[r]);
      mrow[r] = mcur;
      const float alpha = __expf(mprev - mcur);     // 0 on first step
      const float e0 = __expf(s0[r] - mcur);
      const float e1 = __expf(s1[r] - mcur);
      lrow[r] = lrow[r] * alpha + e0 + e1;          // lane-local column sums
      o0[r] *= alpha;
      o1[r] *= alpha;
      const int lr = hb * 8 + r;                    // local row 0..15
      pb[lr * 32 + nl]      = (_Float16)e0;
      pb[lr * 32 + 16 + nl] = (_Float16)e1;
    }
    // P (D-layout) -> A-fragment via wave-private LDS slice (in-order DS)
    v16h pf  = load_a_frag(pb, 32, 0, 0);
    v16h vf0 = load_b_frag(VT, Nq, 0,  j0);   // B^T rows = V^T rows (d), K = keys
    v16h vf1 = load_b_frag(VT, Nq, 16, j0);
    o0 = wmma_f16(pf, vf0, o0);
    o1 = wmma_f16(pf, vf1, o1);
  }

  // finalize: l = sum over the 16 lanes of the half, then O /= l
  #pragma unroll
  for (int r = 0; r < 8; ++r) {
    float l = lrow[r];
    #pragma unroll
    for (int off = 1; off < 16; off <<= 1) l += __shfl_xor(l, off, 32);
    const float inv = 1.0f / l;
    const int n = row0 + hb * 8 + r;
    const size_t base = ((size_t)bp * Nq + n) * Cq + h * Dq;
    ao[base + nl]      = (_Float16)(o0[r] * inv);
    ao[base + 16 + nl] = (_Float16)(o1[r] * inv);
  }
}

// ---------------------------------------------------------------------------
// Stage 3: output projection.  out[t,o] = sum_c a[t,c]*w_proj[o,c] + b[o]
// ---------------------------------------------------------------------------
__global__ void __launch_bounds__(32)
gemm_proj_kernel(const _Float16* __restrict__ ah,
                 const _Float16* __restrict__ wh,
                 const float*   __restrict__ bias,
                 float* __restrict__ out) {
  const int row0 = blockIdx.x * 32;
  const int col0 = blockIdx.y * 64;
  v8f acc[2][4];
  #pragma unroll
  for (int i = 0; i < 2; ++i)
    #pragma unroll
    for (int j = 0; j < 4; ++j) acc[i][j] = (v8f){};

  for (int kk = 0; kk < Cq; kk += 32) {
    v16h a0 = load_a_frag(ah, Cq, row0,      kk);
    v16h a1 = load_a_frag(ah, Cq, row0 + 16, kk);
    v16h b0 = load_b_frag(wh, Cq, col0,      kk);
    v16h b1 = load_b_frag(wh, Cq, col0 + 16, kk);
    v16h b2 = load_b_frag(wh, Cq, col0 + 32, kk);
    v16h b3 = load_b_frag(wh, Cq, col0 + 48, kk);
    acc[0][0] = wmma_f16(a0, b0, acc[0][0]);
    acc[0][1] = wmma_f16(a0, b1, acc[0][1]);
    acc[0][2] = wmma_f16(a0, b2, acc[0][2]);
    acc[0][3] = wmma_f16(a0, b3, acc[0][3]);
    acc[1][0] = wmma_f16(a1, b0, acc[1][0]);
    acc[1][1] = wmma_f16(a1, b1, acc[1][1]);
    acc[1][2] = wmma_f16(a1, b2, acc[1][2]);
    acc[1][3] = wmma_f16(a1, b3, acc[1][3]);
  }

  const int lane = threadIdx.x & 31;
  const int nl = lane & 15, hb = lane >> 4;
  #pragma unroll
  for (int i = 0; i < 2; ++i) {
    #pragma unroll
    for (int j = 0; j < 4; ++j) {
      const int o = col0 + j * 16 + nl;
      const float bv = bias[o];
      #pragma unroll
      for (int r = 0; r < 8; ++r) {
        const int t = row0 + i * 16 + hb * 8 + r;
        out[(size_t)t * Cq + o] = acc[i][j][r] + bv;
      }
    }
  }
}

// ---------------------------------------------------------------------------
// Host launcher
// ---------------------------------------------------------------------------
extern "C" void kernel_launch(void* const* d_in, const int* in_sizes, int n_in,
                              void* d_out, int out_size, void* d_ws, size_t ws_size,
                              hipStream_t stream) {
  const float* x     = (const float*)d_in[0];  // [2,32,512,384]
  const float* pos   = (const float*)d_in[1];  // [12,512,512]
  const float* wqkv  = (const float*)d_in[2];  // [1152,384]
  const float* bqkv  = (const float*)d_in[3];  // [1152]
  const float* wproj = (const float*)d_in[4];  // [384,384]
  const float* bproj = (const float*)d_in[5];  // [384]
  float* out = (float*)d_out;

  // Workspace layout (f16 halves), total ~127 MB.
  const size_t NXC = (size_t)TOK * Cq;          // 12,582,912
  _Float16* xh     = (_Float16*)d_ws;           // x in f16
  _Float16* qh     = xh  + NXC;                 // Q  [bph][n][d]  (pre-scaled)
  _Float16* kh     = qh  + NXC;                 // K  [bph][n][d]
  _Float16* vth    = kh  + NXC;                 // V^T[bph][d][n]
  _Float16* ah     = vth + NXC;                 // attn out [t][h*D+d]
  _Float16* wqkvh  = ah  + NXC;                 // 442,368
  _Float16* wprojh = wqkvh + (size_t)C3 * Cq;   // 147,456

  // Stage 0: conversions
  cvt_f32_f16<<<2048, 256, 0, stream>>>(x,     xh,     (int)(NXC / 4));
  cvt_f32_f16<<<64,   256, 0, stream>>>(wqkv,  wqkvh,  C3 * Cq / 4);
  cvt_f32_f16<<<32,   256, 0, stream>>>(wproj, wprojh, Cq * Cq / 4);

  // Stage 1: QKV projection (M=32768, N=1152, K=384)
  gemm_qkv_kernel<<<dim3(TOK / 32, C3 / 64), 32, 0, stream>>>(
      xh, wqkvh, bqkv, qh, kh, vth);

  // Stage 2: attention (768 heads x 4 row-slabs)
  attn_kernel<<<NHEADS * 4, 256, 0, stream>>>(qh, kh, vth, pos, ah);

  // Stage 3: output projection (M=32768, N=384, K=384)
  gemm_proj_kernel<<<dim3(TOK / 32, Cq / 64), 32, 0, stream>>>(
      ah, wprojh, bproj, out);
}
